// FlospDepth_46188078301700
// MI455X (gfx1250) — matrix operand order repaired and checked by reading, not compiled, not run on hardware
//
#include <hip/hip_runtime.h>
#include <hip/hip_bf16.h>

typedef __attribute__((ext_vector_type(16))) _Float16 v16h;
typedef __attribute__((ext_vector_type(8)))  _Float16 v8h;
typedef __attribute__((ext_vector_type(8)))  float    v8f;

#define BSZ   4
#define CINCH 256
#define HH    56
#define WW    200
#define MIDC  128
#define DCHN  112
#define GXD   128
#define GYD   128
#define GZD   16
#define HWSP  (HH*WW)        // 11200, divisible by 64
#define NSP   (BSZ*HWSP)     // 44800
#define NTILE 8              // 128 / 16 output-channel tiles
#define CSB   (MIDC/32)      // 4 channel blocks in blocked activation layout

// Blocked activation layout: X[n][cs][pixel][ci], ci in [0,32)
//   addr = ((n*CSB + cs)*HWSP + pixel)*32 + ci

// ---------------------------------------------------------------- SE gate MLP
__global__ void gate_kernel(const float* __restrict__ sps,
                            const float* __restrict__ fc1w, const float* __restrict__ fc1b,
                            const float* __restrict__ fc2w, const float* __restrict__ fc2b,
                            const float* __restrict__ serw, const float* __restrict__ serb,
                            const float* __restrict__ seew, const float* __restrict__ seeb,
                            float* __restrict__ gate) {
  __shared__ float bufA[BSZ*MIDC];
  __shared__ float bufB[BSZ*MIDC];
  int t = threadIdx.x;                 // 512 threads
  int n = t >> 7, c = t & 127;
  float s = sps[n];
  bufA[t] = fmaxf(s * fc1w[c] + fc1b[c], 0.f);
  __syncthreads();
  float acc = fc2b[c];
  for (int k = 0; k < MIDC; ++k) acc += bufA[n*MIDC + k] * fc2w[c*MIDC + k];
  bufB[t] = acc;
  __syncthreads();
  acc = serb[c];
  for (int k = 0; k < MIDC; ++k) acc += bufB[n*MIDC + k] * serw[c*MIDC + k];
  bufA[t] = fmaxf(acc, 0.f);
  __syncthreads();
  acc = seeb[c];
  for (int k = 0; k < MIDC; ++k) acc += bufA[n*MIDC + k] * seew[c*MIDC + k];
  gate[t] = 1.f / (1.f + __expf(-acc));
}

// ------------------------------------------------- pack OIHW f32 -> WMMA B frags
// dst layout: [ntile][kstep][lane][16] halves, k = tap*cin + ci
__global__ void pack_conv_w(const float* __restrict__ src, _Float16* __restrict__ dst,
                            int cin, int ksteps) {
  int idx = blockIdx.x * 256 + threadIdx.x;
  int total = NTILE * ksteps * 512;
  if (idx >= total) return;
  int e    = idx & 15;
  int lane = (idx >> 4) & 31;
  int q    = idx >> 9;
  int ks   = q % ksteps;
  int nt   = q / ksteps;
  int nn   = nt*16 + (lane & 15);
  int kb   = (lane < 16) ? 0 : 16;
  int k    = ks*32 + kb + e;
  int tap  = k / cin, ci = k % cin;
  int ky = tap / 3, kx = tap % 3;
  dst[idx] = (_Float16)src[((nn*cin + ci)*3 + ky)*3 + kx];
}

// pack dp_w [112][128] -> [8][4][32][16], zero-pad d>=112
__global__ void pack_dp_w(const float* __restrict__ dpw, _Float16* __restrict__ dst) {
  int idx = blockIdx.x * 256 + threadIdx.x;   // total 8*4*512 = 16384
  int e    = idx & 15;
  int lane = (idx >> 4) & 31;
  int q    = idx >> 9;
  int ks   = q & 3;
  int nt   = q >> 2;
  int d    = nt*16 + (lane & 15);
  int kb   = (lane < 16) ? 0 : 16;
  int k    = ks*32 + kb + e;
  float v  = (d < DCHN) ? dpw[d*MIDC + k] : 0.f;
  dst[idx] = (_Float16)v;
}

// ------------------------------------------------------- implicit-GEMM 3x3 conv
// Block = 256 thr (8 waves). M-tile = 64 rows, N = 128 channels.
// Wave w owns N-tile w; 4 M-subtiles -> 4 WMMA per K-step per wave.
// Double-buffered LDS staging; outputs in channel-blocked f16 layout.
// MODE 0: reduce conv (NCHW f32 src, bias+affine+relu+gate)
// MODE 1: bb conv1    (blocked f16 src, affine+relu)
// MODE 2: bb conv2    (blocked f16 src, affine+residual+relu)
template<int MODE, int CINT>
__global__ __launch_bounds__(256)
void conv3x3_wmma(const void* __restrict__ srcv,
                  const _Float16* __restrict__ wpack,
                  const float* __restrict__ bias, const float* __restrict__ scl,
                  const float* __restrict__ sft, const float* __restrict__ gate,
                  const _Float16* __restrict__ resid, _Float16* __restrict__ out) {
  constexpr int CSTEPS = CINT / 32;
  constexpr int KSTEPS = 9 * CSTEPS;
  __shared__ __align__(16) _Float16 As[2][64*32];   // 2 x 4 KB
  const int tid  = threadIdx.x;
  const int wave = tid >> 5, lane = tid & 31;
  const int sBase = blockIdx.x * 64;

  const int n    = sBase / HWSP;       // uniform (HWSP % 64 == 0)
  const int sRem = sBase - n*HWSP;

  // staging roles
  // MODE 0 (strided f32): thread -> row mS0 = tid&63, 8 channels (tid>>6)*8
  const int mS0 = tid & 63;
  const int cS0 = (tid >> 6) * 8;
  const int rem0 = sRem + mS0;
  const int y0 = rem0 / WW, x0 = rem0 - y0*WW;
  // MODE 1/2 (blocked f16): thread -> row mB = tid>>2, 8 channels (tid&3)*8
  const int mB = tid >> 2;
  const int cB = (tid & 3) * 8;
  const int remB = sRem + mB;
  const int yB = remB / WW, xB = remB - yB*WW;

  const _Float16* bptr = wpack + (size_t)wave*KSTEPS*512 + lane*16;
  const int mA = lane & 15;
  const int kb = (lane < 16) ? 0 : 16;

  const float*    srcF = (const float*)srcv;
  const _Float16* srcH = (const _Float16*)srcv;

  auto stage = [&](int ks, int buf) {
    const int tap = ks / CSTEPS;
    const int cs  = ks - tap*CSTEPS;
    const int ky = tap/3 - 1, kx = tap - (tap/3)*3 - 1;
    if (MODE == 0) {
      const int yy = y0 + ky, xx = x0 + kx;
      const bool valid = (yy >= 0) & (yy < HH) & (xx >= 0) & (xx < WW);
      const int gOff = (valid ? (n*CINT*HWSP + yy*WW + xx) : 0) + (cs*32 + cS0)*HWSP;
      v8h h;
      #pragma unroll
      for (int r = 0; r < 8; ++r) {
        float v = srcF[gOff + r*HWSP];
        h[r] = (_Float16)(valid ? v : 0.f);
      }
      *(v8h*)&As[buf][mS0*32 + cS0] = h;
    } else {
      const int yy = yB + ky, xx = xB + kx;
      const bool valid = (yy >= 0) & (yy < HH) & (xx >= 0) & (xx < WW);
      const int p = valid ? (yy*WW + xx) : 0;
      v8h h = *(const v8h*)&srcH[(((n*CSTEPS + cs)*HWSP + p) << 5) + cB];
      #pragma unroll
      for (int r = 0; r < 8; ++r) h[r] = valid ? h[r] : (_Float16)0.f;
      *(v8h*)&As[buf][mB*32 + cB] = h;
    }
  };

  v8f acc[4] = {};
  stage(0, 0);
  for (int ks = 0; ks < KSTEPS; ++ks) {
    __syncthreads();
    if (ks + 1 < KSTEPS) stage(ks + 1, (ks + 1) & 1);
    v16h b = *(const v16h*)bptr;
    __builtin_prefetch((const void*)(bptr + 512), 0, 1);
    bptr += 512;
    const _Float16* abase = &As[ks & 1][0];
    #pragma unroll
    for (int mt = 0; mt < 4; ++mt) {
      const _Float16* ap = abase + (mt*16 + mA)*32 + kb;
      v16h a;
      #pragma unroll
      for (int j = 0; j < 16; ++j) { a[j] = ap[j]; }
      acc[mt] = __builtin_amdgcn_wmma_f32_16x16x32_f16(
          false, a, false, b, (short)0, acc[mt], false, false);
    }
  }

  // epilogue: blocked-layout stores
  const int nOut = wave*16 + (lane & 15);
  const float sc = scl[nOut];
  const float sh = sft[nOut];
  const float bi = (MODE == 0) ? bias[nOut] : 0.f;
  const float gt = (MODE == 0) ? gate[n*MIDC + nOut] : 0.f;
  const int csOut = nOut >> 5, ciOut = nOut & 31;
  const int oBase = (((n*CSB + csOut)*HWSP + sRem) << 5) + ciOut;
  const int mOff = (lane < 16) ? 0 : 8;
  #pragma unroll
  for (int mt = 0; mt < 4; ++mt) {
    #pragma unroll
    for (int r = 0; r < 8; ++r) {
      const int off = oBase + ((mt*16 + mOff + r) << 5);
      float v = (acc[mt][r] + bi) * sc + sh;
      if (MODE == 0)      v = fmaxf(v, 0.f) * gt;
      else if (MODE == 1) v = fmaxf(v, 0.f);
      else                v = fmaxf(v + (float)resid[off], 0.f);
      out[off] = (_Float16)v;
    }
  }
}

// ---------------------------------------------- depth projection (WMMA) + softmax
__global__ __launch_bounds__(256)
void depth_softmax(const _Float16* __restrict__ x16, const _Float16* __restrict__ wpack,
                   const float* __restrict__ dpb, float* __restrict__ depth) {
  __shared__ __align__(16) _Float16 As[2][64*32];  // 2 x 4 KB
  __shared__ float Cs[64*MIDC];                    // 32 KB
  const int tid  = threadIdx.x;
  const int wave = tid >> 5, lane = tid & 31;
  const int sBase = blockIdx.x * 64;
  const int n    = sBase / HWSP;
  const int sRem = sBase - n*HWSP;
  const int mB = tid >> 2;
  const int cB = (tid & 3) * 8;
  const _Float16* bptr = wpack + (size_t)wave*4*512 + lane*16;
  const int mA = lane & 15;
  const int kb = (lane < 16) ? 0 : 16;

  auto stage = [&](int cs, int buf) {
    v8h h = *(const v8h*)&x16[(((n*CSB + cs)*HWSP + sRem + mB) << 5) + cB];
    *(v8h*)&As[buf][mB*32 + cB] = h;
  };

  v8f acc[4] = {};
  stage(0, 0);
  for (int cs = 0; cs < 4; ++cs) {
    __syncthreads();
    if (cs + 1 < 4) stage(cs + 1, (cs + 1) & 1);
    v16h b = *(const v16h*)bptr;
    bptr += 512;
    const _Float16* abase = &As[cs & 1][0];
    #pragma unroll
    for (int mt = 0; mt < 4; ++mt) {
      const _Float16* ap = abase + (mt*16 + mA)*32 + kb;
      v16h a;
      #pragma unroll
      for (int j = 0; j < 16; ++j) { a[j] = ap[j]; }
      acc[mt] = __builtin_amdgcn_wmma_f32_16x16x32_f16(
          false, a, false, b, (short)0, acc[mt], false, false);
    }
  }
  const int nOut = wave*16 + (lane & 15);
  const int mOff = (lane < 16) ? 0 : 8;
  #pragma unroll
  for (int mt = 0; mt < 4; ++mt)
    #pragma unroll
    for (int r = 0; r < 8; ++r)
      Cs[(mt*16 + mOff + r)*MIDC + nOut] = acc[mt][r];
  __syncthreads();
  if (tid < 64) {
    const int m = tid;
    const float* row = &Cs[m*MIDC];
    float mx = -1e30f;
    for (int d = 0; d < DCHN; ++d) mx = fmaxf(mx, row[d] + dpb[d]);
    float sum = 0.f;
    for (int d = 0; d < DCHN; ++d) sum += __expf(row[d] + dpb[d] - mx);
    const float inv = 1.f / sum;
    const int dBase = n*DCHN*HWSP + sRem + m;
    for (int d = 0; d < DCHN; ++d)
      depth[dBase + d*HWSP] = __expf(row[d] + dpb[d] - mx) * inv;
  }
}

// --------------------------------------------------------- trilinear grid sample
__global__ void grid_sample_kernel(const float* __restrict__ depth,
                                   const float* __restrict__ grids,
                                   float* __restrict__ out) {
  int idx = blockIdx.x * 256 + threadIdx.x;       // BSZ*GXD*GYD*GZD = 1048576
  int gz = idx & (GZD-1);
  int gy = (idx >> 4)  & (GYD-1);
  int gx = (idx >> 11) & (GXD-1);
  int n  = idx >> 18;
  const float* g = grids + ((((size_t)n*GXD + gx)*GYD + gy)*GZD + gz)*3; // cam 0
  float ix = (g[0] + 1.f) * (WW  * 0.5f) - 0.5f;
  float iy = (g[1] + 1.f) * (HH  * 0.5f) - 0.5f;
  float iz = (g[2] + 1.f) * (DCHN* 0.5f) - 0.5f;
  float fx0 = floorf(ix), fy0 = floorf(iy), fz0 = floorf(iz);
  int x0 = (int)fx0, y0 = (int)fy0, z0 = (int)fz0;
  float fx = ix - fx0, fy = iy - fy0, fz = iz - fz0;
  const float* base = depth + (size_t)n * DCHN * HWSP;
  float acc = 0.f;
  #pragma unroll
  for (int dz = 0; dz < 2; ++dz)
  #pragma unroll
  for (int dy = 0; dy < 2; ++dy)
  #pragma unroll
  for (int dx = 0; dx < 2; ++dx) {
    int xi = x0 + dx, yi = y0 + dy, zi = z0 + dz;
    float wgt = (dx ? fx : 1.f-fx) * (dy ? fy : 1.f-fy) * (dz ? fz : 1.f-fz);
    if (xi >= 0 && xi < WW && yi >= 0 && yi < HH && zi >= 0 && zi < DCHN)
      acc += base[(zi*HH + yi)*WW + xi] * wgt;
  }
  out[idx] = acc;
}

// --------------------------------------------------------------------- launcher
extern "C" void kernel_launch(void* const* d_in, const int* in_sizes, int n_in,
                              void* d_out, int out_size, void* d_ws, size_t ws_size,
                              hipStream_t stream) {
  const float* img   = (const float*)d_in[0];
  const float* sps   = (const float*)d_in[1];
  const float* grids = (const float*)d_in[2];
  const float* rc_w  = (const float*)d_in[3];
  const float* rc_b  = (const float*)d_in[4];
  const float* rc_s  = (const float*)d_in[5];
  const float* rc_t  = (const float*)d_in[6];
  const float* fc1w  = (const float*)d_in[7];
  const float* fc1b  = (const float*)d_in[8];
  const float* fc2w  = (const float*)d_in[9];
  const float* fc2b  = (const float*)d_in[10];
  const float* serw  = (const float*)d_in[11];
  const float* serb  = (const float*)d_in[12];
  const float* seew  = (const float*)d_in[13];
  const float* seeb  = (const float*)d_in[14];
  const float* bbw1  = (const float*)d_in[15];
  const float* bbs1  = (const float*)d_in[16];
  const float* bbt1  = (const float*)d_in[17];
  const float* bbw2  = (const float*)d_in[18];
  const float* bbs2  = (const float*)d_in[19];
  const float* bbt2  = (const float*)d_in[20];
  const float* dpw   = (const float*)d_in[21];
  const float* dpb   = (const float*)d_in[22];
  float* out = (float*)d_out;

  char* wsp = (char*)d_ws;
  auto carve = [&](size_t bytes) {
    char* p = wsp; wsp += (bytes + 255) & ~(size_t)255; return p;
  };
  float*    gate = (float*)   carve((size_t)BSZ*MIDC*4);
  _Float16* wrc  = (_Float16*)carve((size_t)NTILE*72*512*2);
  _Float16* wbb[6];
  for (int i = 0; i < 6; ++i) wbb[i] = (_Float16*)carve((size_t)NTILE*36*512*2);
  _Float16* wdp  = (_Float16*)carve((size_t)NTILE*4*512*2);
  _Float16* x16  = (_Float16*)carve((size_t)BSZ*MIDC*HWSP*2);   // blocked layout
  _Float16* y16  = (_Float16*)carve((size_t)BSZ*MIDC*HWSP*2);   // blocked layout
  float*    dep  = (float*)   carve((size_t)BSZ*DCHN*HWSP*4);

  gate_kernel<<<1, 512, 0, stream>>>(sps, fc1w, fc1b, fc2w, fc2b,
                                     serw, serb, seew, seeb, gate);

  pack_conv_w<<<(NTILE*72*512)/256, 256, 0, stream>>>(rc_w, wrc, CINCH, 72);
  for (int i = 0; i < 3; ++i) {
    pack_conv_w<<<(NTILE*36*512)/256, 256, 0, stream>>>(
        bbw1 + (size_t)i*MIDC*MIDC*9, wbb[2*i],   MIDC, 36);
    pack_conv_w<<<(NTILE*36*512)/256, 256, 0, stream>>>(
        bbw2 + (size_t)i*MIDC*MIDC*9, wbb[2*i+1], MIDC, 36);
  }
  pack_dp_w<<<(NTILE*4*512)/256, 256, 0, stream>>>(dpw, wdp);

  const int nblk = NSP / 64;  // 700 M-tiles of 64 rows
  conv3x3_wmma<0, CINCH><<<nblk, 256, 0, stream>>>(img, wrc,
      rc_b, rc_s, rc_t, gate, nullptr, x16);
  for (int i = 0; i < 3; ++i) {
    conv3x3_wmma<1, MIDC><<<nblk, 256, 0, stream>>>(x16, wbb[2*i],
        nullptr, bbs1 + i*MIDC, bbt1 + i*MIDC, nullptr, nullptr, y16);
    conv3x3_wmma<2, MIDC><<<nblk, 256, 0, stream>>>(y16, wbb[2*i+1],
        nullptr, bbs2 + i*MIDC, bbt2 + i*MIDC, nullptr, x16, x16);
  }
  depth_softmax<<<nblk, 256, 0, stream>>>(x16, wdp, dpb, dep);
  grid_sample_kernel<<<(BSZ*GXD*GYD*GZD)/256, 256, 0, stream>>>(dep, grids, out);
}
